// AttentionBlock_66279935312636
// MI455X (gfx1250) — matrix operand order repaired and checked by reading, not compiled, not run on hardware
//
#include <hip/hip_runtime.h>

typedef __attribute__((ext_vector_type(16))) _Float16 v16h;
typedef __attribute__((ext_vector_type(8)))  float    v8f;
typedef __attribute__((ext_vector_type(8)))  _Float16 h8;
typedef __attribute__((ext_vector_type(4)))  _Float16 h4;

#define WMMA_F16(a, b, c) \
  __builtin_amdgcn_wmma_f32_16x16x32_f16(false, (a), false, (b), (short)0, (c), false, false)

static constexpr int Cc = 512;   // channels
static constexpr int Lc = 4096;  // H*W
static constexpr int CH = 64;    // channels per head

// ---- CDNA5 async global->LDS copy (ASYNCcnt) -----------------------------
// GVS mode: mem_addr = SADDR(u64) + VADDR(i32) ; VDST = LDS byte address.
__device__ __forceinline__ void async_cp16(unsigned lds_addr, const void* sbase, int goff) {
  asm volatile("global_load_async_to_lds_b128 %0, %1, %2"
               :: "v"(lds_addr), "v"(goff), "s"(sbase)
               : "memory");
}
__device__ __forceinline__ void wait_async0() {
  asm volatile("s_wait_asynccnt 0" ::: "memory");
}
__device__ __forceinline__ unsigned lds_addr_of(const void* p) {
  return (unsigned)(uintptr_t)p;  // flat shared ptr: low 32 bits = LDS offset
}

// ---- WMMA fragment loaders from LDS -------------------------------------
// A fragment (16x32 f16, MxK): lane row m=lane&15; elems 0..7 = K k0+8*(lane>>4)..,
// elems 8..15 = K k0+16+8*(lane>>4)..  (two ds_load_b128)
__device__ __forceinline__ v16h frag_a(const _Float16* base, int row0, int k0, int stride) {
  const int lane = threadIdx.x & 31;
  const _Float16* p = base + (size_t)(row0 + (lane & 15)) * stride + k0 + ((lane >> 4) << 3);
  v16h r;
  ((uint4*)&r)[0] = *(const uint4*)(p);
  ((uint4*)&r)[1] = *(const uint4*)(p + 16);
  return r;
}
// B fragment (32x16 f16, KxN), LDS stored [n][k]: lane col n=lane&15;
// elems 0..7 = K k0+8*(lane>>4).., elems 8..15 = K k0+16+8*(lane>>4)..
__device__ __forceinline__ v16h frag_b(const _Float16* base, int col0, int k0, int stride) {
  const int lane = threadIdx.x & 31;
  const _Float16* p = base + (size_t)(col0 + (lane & 15)) * stride + k0 + ((lane >> 4) << 3);
  v16h r;
  ((uint4*)&r)[0] = *(const uint4*)(p);
  ((uint4*)&r)[1] = *(const uint4*)(p + 16);
  return r;
}
__device__ __forceinline__ v8f vzero8() {
  v8f z = {0.f, 0.f, 0.f, 0.f, 0.f, 0.f, 0.f, 0.f};
  return z;
}

// ========================== Kernel 1: GroupNorm ===========================
__global__ __launch_bounds__(256) void k_gnorm(const float* __restrict__ x,
                                               const float* __restrict__ gamma,
                                               const float* __restrict__ beta,
                                               _Float16* __restrict__ hid) {
  const int b = blockIdx.x >> 5;
  const int g = blockIdx.x & 31;
  const size_t base = ((size_t)b * Cc + (size_t)g * 16) * Lc;
  const float* xp = x + base;
  __shared__ float r0[256], r1[256];
  float s = 0.f, sq = 0.f;
  for (int i = threadIdx.x; i < 16384; i += 256) {
    float4 v = ((const float4*)xp)[i];
    s  += v.x + v.y + v.z + v.w;
    sq += v.x * v.x + v.y * v.y + v.z * v.z + v.w * v.w;
  }
  r0[threadIdx.x] = s; r1[threadIdx.x] = sq;
  __syncthreads();
  for (int off = 128; off > 0; off >>= 1) {
    if ((int)threadIdx.x < off) {
      r0[threadIdx.x] += r0[threadIdx.x + off];
      r1[threadIdx.x] += r1[threadIdx.x + off];
    }
    __syncthreads();
  }
  const float mean = r0[0] * (1.f / 65536.f);
  const float var  = r1[0] * (1.f / 65536.f) - mean * mean;
  const float rstd = rsqrtf(var + 1e-5f);
  _Float16* hp = hid + base;
  for (int i = threadIdx.x; i < 16384; i += 256) {
    float4 v = ((const float4*)xp)[i];
    const int c = g * 16 + (i >> 10);
    const float ga = gamma[c] * rstd;
    const float be = beta[c] - mean * ga;
    h4 o;
    o[0] = (_Float16)(v.x * ga + be);
    o[1] = (_Float16)(v.y * ga + be);
    o[2] = (_Float16)(v.z * ga + be);
    o[3] = (_Float16)(v.w * ga + be);
    ((h4*)hp)[i] = o;
  }
}

// =================== Kernel 2: QKV GEMM (M=1536,N=4096,K=512) ============
// Writes Q,K transposed ([bh][l][c], channel-contiguous, h8 stores) and V
// natural ([bh][c][l]) so attention can async-DMA row-contiguous tiles.
__global__ __launch_bounds__(256) void k_qkv(const float* __restrict__ W,
                                             const float* __restrict__ bias,
                                             const _Float16* __restrict__ hid,
                                             _Float16* __restrict__ qT,
                                             _Float16* __restrict__ kT,
                                             _Float16* __restrict__ vv) {
  constexpr int STA = 40;
  __shared__ _Float16 sA[128 * STA];
  __shared__ _Float16 sB[64 * STA];
  const int b  = blockIdx.z;
  const int m0 = blockIdx.y * 128;
  const int n0 = blockIdx.x * 64;
  const int tid = threadIdx.x;
  const int w = tid >> 5, lane = tid & 31;
  const int wm = w >> 1, wn = w & 1;
  const int hlo = (lane >> 4) << 3;  // 0 or 8
  const _Float16* hb = hid + (size_t)b * Cc * Lc;
  v8f acc[2][2];
  for (int i = 0; i < 2; ++i) for (int j = 0; j < 2; ++j) acc[i][j] = vzero8();

  for (int k0 = 0; k0 < Cc; k0 += 32) {
    __syncthreads();
#pragma unroll
    for (int i = 0; i < 16; ++i) {
      const int r = i * 8 + (tid >> 5);
      const int c = tid & 31;
      sA[r * STA + c] = (_Float16)W[(size_t)(m0 + r) * Cc + k0 + c];
    }
#pragma unroll
    for (int i = 0; i < 8; ++i) {
      const int c = i * 4 + (tid >> 6);
      const int l = tid & 63;
      sB[l * STA + c] = hb[(size_t)(k0 + c) * Lc + n0 + l];
    }
    __syncthreads();
    v16h af[2], bf[2];
#pragma unroll
    for (int ms = 0; ms < 2; ++ms) af[ms] = frag_a(sA, wm * 32 + ms * 16, 0, STA);
#pragma unroll
    for (int ns = 0; ns < 2; ++ns) bf[ns] = frag_b(sB, wn * 32 + ns * 16, 0, STA);
#pragma unroll
    for (int ms = 0; ms < 2; ++ms)
#pragma unroll
      for (int ns = 0; ns < 2; ++ns)
        acc[ms][ns] = WMMA_F16(af[ms], bf[ns], acc[ms][ns]);
  }
  // Epilogue: rows [mmb, mmb+16) lie entirely inside one q/k/v segment
#pragma unroll
  for (int ms = 0; ms < 2; ++ms) {
    const int mmb  = m0 + wm * 32 + ms * 16;
    const int head = mmb / 192;
    const int segr = mmb % 192;
    const int seg  = segr >> 6;         // 0=q, 1=k, 2=v
    const int c0   = segr & 63;
    const int bh   = b * 8 + head;
#pragma unroll
    for (int ns = 0; ns < 2; ++ns) {
      const int nn = n0 + wn * 32 + ns * 16 + (lane & 15);
      if (seg < 2) {
        h8 o;
#pragma unroll
        for (int r = 0; r < 8; ++r)
          o[r] = (_Float16)(acc[ms][ns][r] + bias[mmb + r + hlo]);
        _Float16* dst = (seg == 0) ? qT : kT;
        *(h8*)(dst + ((size_t)bh * Lc + nn) * CH + c0 + hlo) = o;   // [bh][l][c]
      } else {
#pragma unroll
        for (int r = 0; r < 8; ++r) {
          const int cc = c0 + r + hlo;
          vv[((size_t)bh * CH + cc) * Lc + nn] =
              (_Float16)(acc[ms][ns][r] + bias[mmb + r + hlo]);     // [bh][c][l]
        }
      }
    }
  }
}

// ==================== Kernel 3: flash attention ===========================
// grid.x = L/128 query strips, grid.y = B*NH (16). 256 threads = 8 waves.
__global__ __launch_bounds__(256) void k_attn(const _Float16* __restrict__ qT,
                                              const _Float16* __restrict__ kT,
                                              const _Float16* __restrict__ vv,
                                              _Float16* __restrict__ aout) {
  constexpr int ST = 72;   // padded row stride in halfs (144 B, 16B aligned)
  __shared__ _Float16 sQ[128 * ST];
  __shared__ _Float16 sK[64 * ST];
  __shared__ _Float16 sV[64 * ST];
  __shared__ _Float16 sP[8 * 16 * ST];
  const int bh = blockIdx.y;
  const int t0 = blockIdx.x * 128;
  const int tid = threadIdx.x;
  const int w = tid >> 5, lane = tid & 31;
  const int hlo = (lane >> 4) << 3;
  const _Float16* qg = qT + ((size_t)bh * Lc + t0) * CH;   // [t][c] rows, 128B each
  const _Float16* kg = kT + (size_t)bh * Lc * CH;          // [s][c]
  const _Float16* vg = vv + (size_t)bh * CH * Lc;          // [c][s]
  const unsigned lQ = lds_addr_of(sQ);
  const unsigned lK = lds_addr_of(sK);
  const unsigned lV = lds_addr_of(sV);

  // Stage Q strip: 128 rows x 128B, async DMA into padded LDS rows
#pragma unroll
  for (int i = 0; i < 4; ++i) {
    const int chunk = i * 256 + tid;          // 0..1023 (16B chunks)
    const int t = chunk >> 3, c8 = chunk & 7;
    async_cp16(lQ + t * 144 + c8 * 16, qg, (t * 64 + c8 * 8) * 2);
  }
  wait_async0();
  __syncthreads();
  const v16h aq0 = frag_a(sQ, w * 16, 0, ST);
  const v16h aq1 = frag_a(sQ, w * 16, 32, ST);

  float mx[8], sm[8];
  v8f acc[4];
#pragma unroll
  for (int r = 0; r < 8; ++r) { mx[r] = -1e30f; sm[r] = 0.f; }
#pragma unroll
  for (int ct = 0; ct < 4; ++ct) acc[ct] = vzero8();

  for (int st = 0; st < Lc / 64; ++st) {
    const int s0 = st * 64;
    __syncthreads();                 // all waves done reading previous K/V tile
#pragma unroll
    for (int i = 0; i < 2; ++i) {
      const int chunk = i * 256 + tid;        // 0..511
      const int r = chunk >> 3, c8 = chunk & 7;
      async_cp16(lK + r * 144 + c8 * 16, kg, ((s0 + r) * 64 + c8 * 8) * 2);  // sK[s][c]
      async_cp16(lV + r * 144 + c8 * 16, vg, (r * Lc + s0 + c8 * 8) * 2);    // sV[c][s]
    }
    wait_async0();
    __syncthreads();

    // S = Q^T K : 16(t) x 64(s), K-dim = 64 channels (two WMMA steps)
    v8f sc[4];
#pragma unroll
    for (int ns = 0; ns < 4; ++ns) {
      sc[ns] = WMMA_F16(aq0, frag_b(sK, ns * 16, 0, ST), vzero8());
      sc[ns] = WMMA_F16(aq1, frag_b(sK, ns * 16, 32, ST), sc[ns]);
    }

    // online softmax; row m = r + 8*(lane>=16) lives across a 16-lane half
    float mnew[8], corr[8], rs[8];
#pragma unroll
    for (int r = 0; r < 8; ++r) {
      float v = -1e30f;
#pragma unroll
      for (int ns = 0; ns < 4; ++ns) v = fmaxf(v, sc[ns][r] * 0.125f);
      v = fmaxf(v, __shfl_xor(v, 1));
      v = fmaxf(v, __shfl_xor(v, 2));
      v = fmaxf(v, __shfl_xor(v, 4));
      v = fmaxf(v, __shfl_xor(v, 8));
      mnew[r] = fmaxf(mx[r], v);
      corr[r] = __expf(mx[r] - mnew[r]);
      mx[r] = mnew[r];
      rs[r] = 0.f;
    }
#pragma unroll
    for (int ns = 0; ns < 4; ++ns)
#pragma unroll
      for (int r = 0; r < 8; ++r) {
        const float p = __expf(sc[ns][r] * 0.125f - mnew[r]);
        sc[ns][r] = p;
        rs[r] += p;
      }
#pragma unroll
    for (int r = 0; r < 8; ++r) {
      float t = rs[r];
      t += __shfl_xor(t, 1);
      t += __shfl_xor(t, 2);
      t += __shfl_xor(t, 4);
      t += __shfl_xor(t, 8);
      sm[r] = sm[r] * corr[r] + t;
    }
#pragma unroll
    for (int ct = 0; ct < 4; ++ct)
#pragma unroll
      for (int r = 0; r < 8; ++r) acc[ct][r] *= corr[r];

    // D-frag -> A-frag relayout of P via per-wave LDS (same-wave DS is in-order)
    _Float16* pp = sP + w * 16 * ST;
#pragma unroll
    for (int ns = 0; ns < 4; ++ns)
#pragma unroll
      for (int r = 0; r < 8; ++r)
        pp[(r + hlo) * ST + ns * 16 + (lane & 15)] = (_Float16)sc[ns][r];

    const v16h ap0 = frag_a(pp, 0, 0, ST);
    const v16h ap1 = frag_a(pp, 0, 32, ST);
#pragma unroll
    for (int ct = 0; ct < 4; ++ct) {
      acc[ct] = WMMA_F16(ap0, frag_b(sV, ct * 16, 0, ST), acc[ct]);
      acc[ct] = WMMA_F16(ap1, frag_b(sV, ct * 16, 32, ST), acc[ct]);
    }
  }

  // a[c][t]: lane's 8 rows are 8 consecutive t -> one 16B store per tile
  _Float16* ab = aout + ((size_t)bh >> 3) * Cc * Lc;  // batch base
  const int head = bh & 7;
#pragma unroll
  for (int ct = 0; ct < 4; ++ct) {
    const int cc = head * CH + ct * 16 + (lane & 15);
    h8 o;
#pragma unroll
    for (int r = 0; r < 8; ++r) o[r] = (_Float16)(acc[ct][r] * (1.f / sm[r]));
    *(h8*)(ab + (size_t)cc * Lc + t0 + w * 16 + hlo) = o;
  }
}

// ============ Kernel 4: proj GEMM + bias + residual (f32 out) =============
__global__ __launch_bounds__(256) void k_proj(const float* __restrict__ W,
                                              const float* __restrict__ bias,
                                              const _Float16* __restrict__ a,
                                              const float* __restrict__ x,
                                              float* __restrict__ out) {
  constexpr int STA = 40;
  __shared__ _Float16 sA[128 * STA];
  __shared__ _Float16 sB[64 * STA];
  const int b  = blockIdx.z;
  const int m0 = blockIdx.y * 128;
  const int n0 = blockIdx.x * 64;
  const int tid = threadIdx.x;
  const int w = tid >> 5, lane = tid & 31;
  const int wm = w >> 1, wn = w & 1;
  const _Float16* ab = a + (size_t)b * Cc * Lc;
  v8f acc[2][2];
  for (int i = 0; i < 2; ++i) for (int j = 0; j < 2; ++j) acc[i][j] = vzero8();

  for (int k0 = 0; k0 < Cc; k0 += 32) {
    __syncthreads();
#pragma unroll
    for (int i = 0; i < 16; ++i) {
      const int r = i * 8 + (tid >> 5);
      const int c = tid & 31;
      sA[r * STA + c] = (_Float16)W[(size_t)(m0 + r) * Cc + k0 + c];
    }
#pragma unroll
    for (int i = 0; i < 8; ++i) {
      const int c = i * 4 + (tid >> 6);
      const int l = tid & 63;
      sB[l * STA + c] = ab[(size_t)(k0 + c) * Lc + n0 + l];
    }
    __syncthreads();
    v16h af[2], bf[2];
#pragma unroll
    for (int ms = 0; ms < 2; ++ms) af[ms] = frag_a(sA, wm * 32 + ms * 16, 0, STA);
#pragma unroll
    for (int ns = 0; ns < 2; ++ns) bf[ns] = frag_b(sB, wn * 32 + ns * 16, 0, STA);
#pragma unroll
    for (int ms = 0; ms < 2; ++ms)
#pragma unroll
      for (int ns = 0; ns < 2; ++ns)
        acc[ms][ns] = WMMA_F16(af[ms], bf[ns], acc[ms][ns]);
  }
  const float* xb = x + (size_t)b * Cc * Lc;
  float* ob = out + (size_t)b * Cc * Lc;
#pragma unroll
  for (int ms = 0; ms < 2; ++ms)
#pragma unroll
    for (int ns = 0; ns < 2; ++ns) {
      const int nn = n0 + wn * 32 + ns * 16 + (lane & 15);
#pragma unroll
      for (int r = 0; r < 8; ++r) {
        const int mm = m0 + wm * 32 + ms * 16 + r + ((lane >> 4) << 3);
        const size_t idx = (size_t)mm * Lc + nn;
        ob[idx] = xb[idx] + acc[ms][ns][r] + bias[mm];
      }
    }
}

// ============================== launcher ==================================
extern "C" void kernel_launch(void* const* d_in, const int* in_sizes, int n_in,
                              void* d_out, int out_size, void* d_ws, size_t ws_size,
                              hipStream_t stream) {
  const float* x      = (const float*)d_in[0];
  const float* norm_w = (const float*)d_in[1];
  const float* norm_b = (const float*)d_in[2];
  const float* qkv_w  = (const float*)d_in[3];
  const float* qkv_b  = (const float*)d_in[4];
  const float* proj_w = (const float*)d_in[5];
  const float* proj_b = (const float*)d_in[6];
  float* out = (float*)d_out;

  char* ws = (char*)d_ws;
  constexpr size_t SEG = (size_t)16 * Lc * CH * sizeof(_Float16);  // 8 MB each
  _Float16* hid = (_Float16*)(ws);
  _Float16* qT  = (_Float16*)(ws + SEG);
  _Float16* kT  = (_Float16*)(ws + 2 * SEG);
  _Float16* vv  = (_Float16*)(ws + 3 * SEG);
  _Float16* ah  = (_Float16*)(ws + 4 * SEG);

  k_gnorm<<<dim3(64), dim3(256), 0, stream>>>(x, norm_w, norm_b, hid);
  k_qkv<<<dim3(Lc / 64, (3 * Cc) / 128, 2), dim3(256), 0, stream>>>(qkv_w, qkv_b, hid, qT, kT, vv);
  k_attn<<<dim3(Lc / 128, 16), dim3(256), 0, stream>>>(qT, kT, vv, ah);
  k_proj<<<dim3(Lc / 64, Cc / 128, 2), dim3(256), 0, stream>>>(proj_w, proj_b, ah, x, out);
}
